// _MLX4Linear_257698038253
// MI455X (gfx1250) — matrix operand order repaired and checked by reading, not compiled, not run on hardware
//
#include <hip/hip_runtime.h>

typedef _Float16 h2   __attribute__((ext_vector_type(2)));
typedef _Float16 h4   __attribute__((ext_vector_type(4)));
typedef _Float16 h8   __attribute__((ext_vector_type(8)));
typedef _Float16 v16h __attribute__((ext_vector_type(16)));
typedef float    v8f  __attribute__((ext_vector_type(8)));
typedef float    f4   __attribute__((ext_vector_type(4)));
typedef int      i4   __attribute__((ext_vector_type(4)));

#define O_DIM  11008
#define I_DIM  4096
#define NGROUP 64
#define B_TOK  256

// ---- pre-pass: x f32 -> f16 into workspace (4 floats / thread) ----
__global__ __launch_bounds__(256)
void xcvt_kernel(const float* __restrict__ x, _Float16* __restrict__ xh)
{
    const int idx = (blockIdx.x * 256 + threadIdx.x) * 4;
    const f4 v = *(const f4*)(x + idx);
    h4 r;
#pragma unroll
    for (int j = 0; j < 4; ++j) r[j] = (_Float16)v[j];
    *(h4*)(xh + idx) = r;
}

// ---- main GEMM: C[256,11008] = X * dequant(Wq)^T + bias ----
// Block: 64(M) x 256(N), 8 waves side by side in N.
// Wave: 64(M) x 32(N) = 4x2 WMMA tiles -> each dequantized B-frag feeds 4 WMMAs.
template<bool XF16>
__global__ __launch_bounds__(256)
void mlx4_wmma_kernel(const float* __restrict__ x32,
                      const _Float16* __restrict__ x16,
                      const int* __restrict__ wq,
                      const _Float16* __restrict__ scales,
                      const _Float16* __restrict__ zeros,
                      const _Float16* __restrict__ bias,
                      float* __restrict__ out)
{
    const int tid  = threadIdx.x;
    const int lane = tid & 31;
    const int wave = tid >> 5;      // 0..7, N direction
    const int l16  = lane & 15;
    const int lh   = lane >> 4;

    const int nb = blockIdx.x * 256 + wave * 32;  // wave N base (32 outs)
    const int mb = blockIdx.y * 64;               // block/wave M base (64 tokens)

    v8f acc[4][2];
#pragma unroll
    for (int t = 0; t < 4; ++t)
#pragma unroll
        for (int tn = 0; tn < 2; ++tn)
#pragma unroll
            for (int e = 0; e < 8; ++e)
                acc[t][tn][e] = 0.0f;

    const float*    x32row[4];
    const _Float16* x16row[4];
#pragma unroll
    for (int t = 0; t < 4; ++t) {
        const size_t row = (size_t)(mb + t * 16 + l16) * I_DIM;
        if (XF16) x16row[t] = x16 + row;
        else      x32row[t] = x32 + row;
    }

    const int*      wrow[2];
    const _Float16* srow[2];
    const _Float16* zrow[2];
#pragma unroll
    for (int tn = 0; tn < 2; ++tn) {
        const int n = nb + tn * 16 + l16;
        wrow[tn] = wq     + (size_t)n * (I_DIM / 2);
        srow[tn] = scales + (size_t)n * NGROUP;
        zrow[tn] = zeros  + (size_t)n * NGROUP;
    }

    for (int g = 0; g < NGROUP; ++g) {
        // per-group scale / adjusted zero (z' = z - 1024*s, for the 0x6400 nibble trick)
        h2 s2[2], z2[2];
#pragma unroll
        for (int tn = 0; tn < 2; ++tn) {
            const float s = (float)srow[tn][g];
            const float z = (float)zrow[tn][g];
            const _Float16 sh = (_Float16)s;
            const _Float16 zh = (_Float16)(z - 1024.0f * s);
            s2[tn][0] = sh; s2[tn][1] = sh;
            z2[tn][0] = zh; z2[tn][1] = zh;
        }

#pragma unroll
        for (int kk = 0; kk < 2; ++kk) {
            const int kb = g * 64 + kk * 32;

            // ---- A fragments (16x32 f16): lane m=l16; elems 0..7 = k(kb+8lh ..+7),
            // elems 8..15 = k(kb+16+8lh ..+7) -> two aligned 16B runs.
            v16h afrag[4];
#pragma unroll
            for (int t = 0; t < 4; ++t) {
                if (XF16) {
                    const _Float16* pa = x16row[t] + kb + 8 * lh;
                    const h8 lo = *(const h8*)(pa);
                    const h8 hi = *(const h8*)(pa + 16);
#pragma unroll
                    for (int j = 0; j < 8; ++j) {
                        afrag[t][j]     = lo[j];
                        afrag[t][8 + j] = hi[j];
                    }
                } else {
                    const float* pa = x32row[t] + kb + 8 * lh;
                    const f4 f0 = *(const f4*)(pa);
                    const f4 f1 = *(const f4*)(pa + 4);
                    const f4 f2 = *(const f4*)(pa + 16);
                    const f4 f3 = *(const f4*)(pa + 20);
#pragma unroll
                    for (int j = 0; j < 4; ++j) {
                        afrag[t][j]      = (_Float16)f0[j];
                        afrag[t][4 + j]  = (_Float16)f1[j];
                        afrag[t][8 + j]  = (_Float16)f2[j];
                        afrag[t][12 + j] = (_Float16)f3[j];
                    }
                }
            }

            // ---- B fragments (32x16 f16): lane n=l16; elems 0..7 = k(kb+8lh ..+7),
            // elems 8..15 = k(kb+16+8lh ..+7) -> two runs of 4 int32s of weight_q.
#pragma unroll
            for (int tn = 0; tn < 2; ++tn) {
                const int* pw = wrow[tn] + ((kb + 8 * lh) >> 1);
                const i4 wa = *(const i4*)(pw);
                const i4 wb = *(const i4*)(pw + 8);
                const unsigned bytes[8] = {
                    (unsigned)wa[0], (unsigned)wa[1], (unsigned)wa[2], (unsigned)wa[3],
                    (unsigned)wb[0], (unsigned)wb[1], (unsigned)wb[2], (unsigned)wb[3]
                };
                union { v16h h; unsigned u[8]; } bf;
#pragma unroll
                for (int v = 0; v < 8; ++v) {
                    const unsigned b = bytes[v] & 0xFFu;
                    union { unsigned u; h2 h; } q;
                    q.u = 0x64006400u | (b & 0xFu) | ((b & 0xF0u) << 12); // {1024+qlo,1024+qhi}
                    union { h2 h; unsigned u; } r;
                    r.h = q.h * s2[tn] + z2[tn];   // v_pk_fma_f16
                    bf.u[v] = r.u;
                }
                // one dequantized B-frag feeds 4 WMMAs (4 M tiles)
#pragma unroll
                for (int t = 0; t < 4; ++t)
                    acc[t][tn] = __builtin_amdgcn_wmma_f32_16x16x32_f16(
                        false, afrag[t], false, bf.h,
                        (short)0, acc[t][tn], false, false);
            }
        }
    }

    // ---- epilogue: bias add + store (C layout: n=l16, m = vgpr + 8*lh)
    float bv[2];
#pragma unroll
    for (int tn = 0; tn < 2; ++tn)
        bv[tn] = (float)bias[nb + tn * 16 + l16];

#pragma unroll
    for (int t = 0; t < 4; ++t) {
#pragma unroll
        for (int v = 0; v < 8; ++v) {
            const int m = mb + t * 16 + 8 * lh + v;
            float* orow = out + (size_t)m * O_DIM;
#pragma unroll
            for (int tn = 0; tn < 2; ++tn)
                orow[nb + tn * 16 + l16] = acc[t][tn][v] + bv[tn];
        }
    }
}

extern "C" void kernel_launch(void* const* d_in, const int* in_sizes, int n_in,
                              void* d_out, int out_size, void* d_ws, size_t ws_size,
                              hipStream_t stream) {
    (void)in_sizes; (void)n_in; (void)out_size;
    const float*    x  = (const float*)d_in[0];
    const int*      wq = (const int*)d_in[1];
    const _Float16* sc = (const _Float16*)d_in[2];
    const _Float16* zr = (const _Float16*)d_in[3];
    const _Float16* bi = (const _Float16*)d_in[4];
    float* out = (float*)d_out;

    const dim3 grid(O_DIM / 256, B_TOK / 64);   // (43, 4)
    const dim3 block(256);
    const size_t need = (size_t)B_TOK * I_DIM * sizeof(_Float16);  // 2 MB

    if (ws_size >= need) {
        _Float16* xh = (_Float16*)d_ws;
        xcvt_kernel<<<(B_TOK * I_DIM / 4) / 256, 256, 0, stream>>>(x, xh);
        mlx4_wmma_kernel<true><<<grid, block, 0, stream>>>(x, xh, wq, sc, zr, bi, out);
    } else {
        mlx4_wmma_kernel<false><<<grid, block, 0, stream>>>(x, nullptr, wq, sc, zr, bi, out);
    }
}
